// NonDynamicToepliztMultihead_50672024158822
// MI455X (gfx1250) — matrix operand order repaired and checked by reading, not compiled, not run on hardware
//
#include <hip/hip_runtime.h>

typedef __attribute__((ext_vector_type(16))) __bf16 v16bf;
typedef __attribute__((ext_vector_type(8)))  float  v8f;

#define Bn 8
#define Hn 16
#define Nn 2048
#define Dn 64
#define NTHREADS 512
#define APAD 32                 // covers t down to -31 in the diagonal K-block
#define ALEN (Nn + APAD)        // reversed tap array length (2080)
#define XROWS 16                // rows 8..15 are zero padding (kills divergence)
#define XSTRIDE (Nn + 8)        // +8 elements (16B) row pad: bank-conflict-free

union BFrag {
  v16bf v;
  uint4 q[2];
};

__device__ __forceinline__ v16bf load_a_frag(const __bf16* ap) {
  // Two ascending 8-element runs -> two (unaligned) ds_load_b128, no swaps.
  v16bf f;
  #pragma unroll
  for (int e = 0; e < 8; ++e)  f[e] = ap[e];        // k = half*8 + e
  #pragma unroll
  for (int e = 8; e < 16; ++e) f[e] = ap[8 + e];    // k = half*8 + e + 8
  return f;
}

__global__ __launch_bounds__(NTHREADS)
void toeplitz_wmma_kernel(const float* __restrict__ xg,
                          const float* __restrict__ posg,
                          const float* __restrict__ zerog,
                          const float* __restrict__ gammag,
                          float* __restrict__ outg) {
  // REVERSED taps: ar[s] = a[Nn-1-s]; s > Nn-1 -> t < 0 -> 0 (causal pad).
  __shared__ __align__(16) __bf16 ar[ALEN];
  __shared__ __align__(16) __bf16 x_lds[XROWS][XSTRIDE];

  const int h   = blockIdx.x >> 6;   // / 64
  const int d   = blockIdx.x & 63;
  const int tid = threadIdx.x;

  // ---- Build reversed taps: t = 2047 - s; a[0]=zero term, a[t]=g^t*pos[t-1] ----
  const float g = gammag[h * Dn + d];
  for (int s = tid; s < ALEN; s += NTHREADS) {
    int t = (Nn - 1) - s;
    float v;
    if (t < 0)       v = 0.0f;
    else if (t == 0) v = zerog[h * Dn + d];
    else             v = __powf(g, (float)t) *
                         posg[((size_t)h * (Nn - 1) + (t - 1)) * Dn + d];
    ar[s] = (__bf16)v;
  }
  // ---- Stage x[:, h, :, d] as bf16; rows 8..15 zero (B-matrix column pad) ----
  for (int idx = tid; idx < XROWS * Nn; idx += NTHREADS) {
    int b = idx >> 11;          // / 2048
    int j = idx & (Nn - 1);
    float v = (b < Bn) ? xg[(((size_t)b * Hn + h) * Nn + j) * Dn + d] : 0.0f;
    x_lds[b][j] = (__bf16)v;
  }
  __syncthreads();

  // Scalarize the wave index so loop bounds live in SGPRs (scalar branches,
  // no per-iteration exec masking).
  const int wave = __builtin_amdgcn_readfirstlane(tid >> 5);   // 16 waves
  const int lane = tid & 31;
  const int m    = lane & 15;   // A row / B,C,D column within tile
  const int half = lane >> 4;   // lane half selects K-range / M-offset

  // Wave w handles row-block pairs (Ilo, Ihi) = (w+16r, w+16r+64), r = 0..3.
  // B fragments depend only on j0 -> shared between the pair's two WMMAs.
  #pragma unroll 1
  for (int r = 0; r < 4; ++r) {
    const int Ilo = wave + 16 * r;
    const int Ihi = Ilo + 64;
    v8f accL = {};
    v8f accH = {};
    const int sbL = (Nn - 1) - (16 * Ilo + m) + half * 8;  // reversed-tap base
    const int sbH = (Nn - 1) - (16 * Ihi + m) + half * 8;
    const int jmaxL = 16 * Ilo + 16;   // causal bounds (uniform -> SGPR)
    const int jmaxH = 16 * Ihi + 16;

    int j0 = 0;
    // Shared phase: one B load feeds two WMMAs.
    for (; j0 < jmaxL; j0 += 32) {
      BFrag bf;
      const uint4* xr = reinterpret_cast<const uint4*>(&x_lds[m][j0 + half * 8]);
      bf.q[0] = xr[0];
      bf.q[1] = xr[2];
      v16bf aL = load_a_frag(&ar[sbL + j0]);
      v16bf aH = load_a_frag(&ar[sbH + j0]);
      accL = __builtin_amdgcn_wmma_f32_16x16x32_bf16(
          false, aL, false, bf.v, (short)0, accL, false, false);
      accH = __builtin_amdgcn_wmma_f32_16x16x32_bf16(
          false, aH, false, bf.v, (short)0, accH, false, false);
    }
    // Tail phase: finish the high block alone.
    for (; j0 < jmaxH; j0 += 32) {
      BFrag bf;
      const uint4* xr = reinterpret_cast<const uint4*>(&x_lds[m][j0 + half * 8]);
      bf.q[0] = xr[0];
      bf.q[1] = xr[2];
      v16bf aH = load_a_frag(&ar[sbH + j0]);
      accH = __builtin_amdgcn_wmma_f32_16x16x32_bf16(
          false, aH, false, bf.v, (short)0, accH, false, false);
    }

    // C/D layout: element q -> row M = q + 8*half, column N = lane&15 (= b).
    if (m < Bn) {
      float* opL = outg + (((size_t)m * Hn + h) * Nn + 16 * Ilo + 8 * half) * Dn + d;
      float* opH = outg + (((size_t)m * Hn + h) * Nn + 16 * Ihi + 8 * half) * Dn + d;
      #pragma unroll
      for (int q = 0; q < 8; ++q) opL[(size_t)q * Dn] = accL[q];
      #pragma unroll
      for (int q = 0; q < 8; ++q) opH[(size_t)q * Dn] = accH[q];
    }
  }
}

extern "C" void kernel_launch(void* const* d_in, const int* in_sizes, int n_in,
                              void* d_out, int out_size, void* d_ws, size_t ws_size,
                              hipStream_t stream) {
  const float* x     = (const float*)d_in[0];
  const float* pos   = (const float*)d_in[1];
  const float* zero  = (const float*)d_in[2];
  const float* gamma = (const float*)d_in[3];
  float* out = (float*)d_out;

  dim3 grid(Hn * Dn);      // one workgroup per (h, d)
  dim3 block(NTHREADS);    // 16 waves of 32
  hipLaunchKernelGGL(toeplitz_wmma_kernel, grid, block, 0, stream,
                     x, pos, zero, gamma, out);
}
